// MelSpectrogramLayer_7722351198625
// MI455X (gfx1250) — matrix operand order repaired and checked by reading, not compiled, not run on hardware
//
#include <hip/hip_runtime.h>

// ---------------------------------------------------------------------------
// MelSpectrogram (Gaussian-window Gabor transform) for MI455X / gfx1250.
// Truncated-support windowed DFT as f16 WMMA GEMMs with f32 accumulation,
// fused power-spectrum + mel contraction (also WMMA).  x-window staged into
// LDS via the Tensor Data Mover (TENSOR_LOAD_TO_LDS + s_wait_tensorcnt).
// ---------------------------------------------------------------------------

typedef _Float16 h16;
typedef __attribute__((ext_vector_type(16))) _Float16 v16h;
typedef __attribute__((ext_vector_type(8)))  _Float16 v8h;
typedef __attribute__((ext_vector_type(8)))  float    v8f;
typedef __attribute__((ext_vector_type(4)))  unsigned int v4u;
typedef __attribute__((ext_vector_type(4)))  int v4i;
typedef __attribute__((ext_vector_type(8)))  int v8i;

#if __has_include(<hip/amd_detail/amd_gfx1250_TDM.h>)
#define TDM_6ARG 1          // amdgpu-toolchain (clang-23 / therock-10.0 headers)
#else
#define TDM_6ARG 0          // ROCm 7.2 (clang-22): 5-arg builtin
#endif

#define BATCH   32
#define NPTS    8192
#define TOUT    513
#define TT      33          // 16-frame t-tiles (33*16 = 528 >= 513)
#define NFREQ   4097
#define FTILES  258         // 16-wide freq tiles (pad 4097 -> 4128)
#define FCHUNKS 129         // 32-wide freq chunks
#define NMELS   128
#define KW      3840        // truncated window support per t-tile (multiple of 32)
#define KBLK    120         // KW/32
#define KABS    256         // 8192/32 absolute k-blocks
#define NWAVES  8

// workspace layout (bytes). Requires ws_size >= ~138 MB.
#define XC_OFF   0u
#define COS_OFF  (1u << 20)
#define TRIG_ELE (KABS * FTILES * 512)              // 33,816,576 halves per matrix
#define SIN_OFF  (COS_OFF + TRIG_ELE * 2u)
#define FB_OFF   (SIN_OFF + TRIG_ELE * 2u)
#define FB_ELE   (FCHUNKS * 8 * 512)

// dynamic LDS layout (bytes)
#define L_XWIN   0
#define L_A      (KW * 4)                            // 15360
#define L_MEL    (L_A + KBLK * 512 * 2)              // 15360 + 122880 = 138240
#define L_SPEC   (L_MEL + 16 * NMELS * 4)            // +8192 = 146432
#define SMEM_BYTES (L_SPEC + NWAVES * 512 * 2)       // +8192 = 154624

static __device__ inline v8f vzero8() {
    v8f z;
#pragma unroll
    for (int i = 0; i < 8; ++i) z[i] = 0.0f;
    return z;
}

static __device__ inline v8f wmma_f16(v16h a, v16h b, v8f c) {
    // v_wmma_f32_16x16x32_f16  D = A(16x32 f16) * B(32x16 f16) + C(16x16 f32)
    return __builtin_amdgcn_wmma_f32_16x16x32_f16(false, a, false, b,
                                                  (short)0, c, false, false);
}

// Issue a 1D TENSOR_LOAD_TO_LDS: LDS[lds_addr .. +tile0*4) = xc[start .. ),
// with TDM OOB semantics zero-filling reads past tensor_dim0.
static __device__ inline void tdm_load_1d(unsigned int lds_addr,
                                          const float* gsrc,
                                          unsigned int tdim0,
                                          unsigned int tile0) {
    const unsigned long long ga = (unsigned long long)(const void*)gsrc;
    // D# group0: count=1 | lds_addr | global_addr[56:0] | type=2
    const v4u g0 = { 1u,
                     lds_addr,
                     (unsigned int)ga,
                     (unsigned int)((ga >> 32) & 0x01FFFFFFu) | 0x80000000u };
    // D# group1: data_size=2 (4B); tensor_dim0; tensor_dim1=1;
    //            tile_dim0=tile0; tile_dim1/2=0; dim0_stride=tensor_dim0.
    const v8i g1 = { (int)0x00020000u,
                     (int)((tdim0 & 0xFFFFu) << 16),
                     (int)(((tdim0 >> 16) & 0xFFFFu) | (1u << 16)),
                     (int)(tile0 << 16),
                     0,
                     (int)tdim0,
                     0, 0 };
    const v4i z4 = { 0, 0, 0, 0 };
#if TDM_6ARG
    const v8i z8 = { 0, 0, 0, 0, 0, 0, 0, 0 };
    __builtin_amdgcn_tensor_load_to_lds(g0, g1, z4, z4, z8, 0);
#else
    __builtin_amdgcn_tensor_load_to_lds(g0, g1, z4, z4, 0);
#endif
}

// ---------------------------------------------------------------------------
// Prep kernel 1: per-row mean removal.
// ---------------------------------------------------------------------------
__global__ void melgab_center(const float* __restrict__ x, float* __restrict__ xc) {
    __shared__ float red[256];
    const int b = blockIdx.x, t = threadIdx.x;
    float s = 0.f;
    for (int i = t; i < NPTS; i += 256) s += x[b * NPTS + i];
    red[t] = s;
    __syncthreads();
    for (int w = 128; w > 0; w >>= 1) {
        if (t < w) red[t] += red[t + w];
        __syncthreads();
    }
    const float mean = red[0] * (1.0f / NPTS);
    for (int i = t; i < NPTS; i += 256) xc[b * NPTS + i] = x[b * NPTS + i] - mean;
}

// ---------------------------------------------------------------------------
// Prep kernel 2: cos/sin DFT matrices, pre-swizzled into WMMA B-operand
// layout.  B tile (kb,ft): 512 f16; lane l covers f = ft*16+(l&15),
// element j covers K = j + (l>=16 ? 16 : 0).  Exact: cos(2*pi*((n*f)&8191)/8192).
// ---------------------------------------------------------------------------
__global__ void melgab_trig(h16* __restrict__ cosB, h16* __restrict__ sinB) {
    const int kb = blockIdx.x / FTILES;
    const int ft = blockIdx.x % FTILES;
    for (int e = threadIdx.x; e < 512; e += 256) {
        const int l = e >> 4, j = e & 15;
        const int n = kb * 32 + j + ((l >= 16) ? 16 : 0);
        const int f = ft * 16 + (l & 15);
        const int ph = (n * f) & (NPTS - 1);
        const float ang = (float)ph * 7.6699039394282e-4f;   // 2*pi/8192
        float s, c;
        __sincosf(ang, &s, &c);
        const int o = blockIdx.x * 512 + e;
        cosB[o] = (h16)c;
        sinB[o] = (h16)s;   // sign of sin is irrelevant: spec = re^2 + im^2
    }
}

// ---------------------------------------------------------------------------
// Prep kernel 3: HTK mel filterbank (torchaudio melscale_fbanks, norm=None),
// pre-swizzled into WMMA B layout.  Tile (fc,mt): lane l -> m = mt*16+(l&15),
// element j -> f = fc*32 + j + (l>=16 ? 16 : 0).  fb = 0 for padded f > 4096.
// ---------------------------------------------------------------------------
static __device__ inline float melgab_mel2hz(float m) {
    return 700.f * (__powf(10.f, m * (1.f / 2595.f)) - 1.f);
}

__global__ void melgab_fb(h16* __restrict__ fbsw) {
    const int idx = blockIdx.x * 256 + threadIdx.x;
    if (idx >= FB_ELE) return;
    const int tile = idx >> 9, r = idx & 511, l = r >> 4, j = r & 15;
    const int fc = tile >> 3, mt = tile & 7;
    const int f = fc * 32 + j + ((l >= 16) ? 16 : 0);
    const int m = mt * 16 + (l & 15);
    float val = 0.f;
    if (f <= NPTS / 2) {
        const float mel_max = 2595.f * log10f(1.f + 8000.f / 700.f);
        const float p0 = melgab_mel2hz(mel_max * (float)(m)     * (1.f / 129.f));
        const float p1 = melgab_mel2hz(mel_max * (float)(m + 1) * (1.f / 129.f));
        const float p2 = melgab_mel2hz(mel_max * (float)(m + 2) * (1.f / 129.f));
        const float freq = (float)f * (8000.f / 4096.f);
        const float down = (freq - p0) / (p1 - p0);
        const float up   = (p2 - freq) / (p2 - p1);
        val = fmaxf(0.f, fminf(down, up));
    }
    fbsw[idx] = (h16)val;
}

// ---------------------------------------------------------------------------
// Main kernel: one 8-wave workgroup per (batch b, 16-frame t-tile).
//   Phase 1: TDM DMA of the x window into LDS (wave 0 issues; OOB -> zeros).
//   Phase 2: build windowed A matrix (16 frames x KW) in LDS, f16, A layout.
//   Phase 3: each wave owns 32-freq chunks; K loop of v_wmma_f32_16x16x32_f16
//            for Re/Im; fuse |.|^2 and mel contraction (8 more WMMAs) keeping
//            the 16x128 mel accumulator in registers.
//   Phase 4: cross-wave LDS reduction and store.
// ---------------------------------------------------------------------------
__global__ void __launch_bounds__(256)
melgab_main(const float* __restrict__ xc, const h16* __restrict__ cosB,
            const h16* __restrict__ sinB, const h16* __restrict__ fbsw,
            const float* __restrict__ lambd, float* __restrict__ out) {
    extern __shared__ char smem[];
    float* xwin  = (float*)(smem + L_XWIN);
    h16*   Ash   = (h16*)  (smem + L_A);
    float* melb  = (float*)(smem + L_MEL);
    h16*   specb = (h16*)  (smem + L_SPEC);

    const int b  = blockIdx.x & 31;
    const int tb = blockIdx.x >> 5;
    const int t0 = tb * 16;
    const int nbase = 256 * tb - 1792;          // multiple of 32; support union
    const int tid = threadIdx.x;
    const float inv_s = 1.0f / fabsf(lambd[0]);

    // phase 1: head zeros (region TDM skips) + mel buffer zero, then TDM load.
    const int head = (nbase < 0) ? -nbase : 0;  // leading out-of-signal samples
    for (int i = tid; i < head; i += 256) xwin[i] = 0.f;
    for (int i = tid; i < 16 * NMELS; i += 256) melb[i] = 0.f;

    if (tid < 32) {                             // wave 0 owns the DMA
        const int start = (nbase > 0) ? nbase : 0;
        // low 32 bits of a generic LDS pointer == LDS byte address
        const unsigned int lds_addr =
            (unsigned int)(unsigned long long)(void*)(xwin + head);
        tdm_load_1d(lds_addr, xc + (size_t)b * NPTS + start,
                    (unsigned int)(NPTS - start),    // tail OOB reads -> zeros
                    (unsigned int)(KW - head));
        __builtin_amdgcn_s_wait_tensorcnt(0);   // TENSORcnt is per-wave
    }
    __syncthreads();                            // publish LDS tile to all waves

    // phase 2: A matrix, f16, pre-swizzled A-operand layout per 32-K block:
    //   element (l,j) of block kb -> A[M = l&15][K = j + (l>=16)*16]
    for (int e = tid; e < KBLK * 512; e += 256) {
        const int kb = e >> 9, r = e & 511, l = r >> 4, j = r & 15;
        const int K = j + ((l >= 16) ? 16 : 0);
        const int i = kb * 32 + K;
        const float c = (float)((t0 + (l & 15)) * 16);
        const float d = ((float)(nbase + i) - c) * inv_s;
        Ash[e] = (h16)(xwin[i] * __expf(-0.5f * d * d));
    }
    __syncthreads();

    const int wave = tid >> 5, lane = tid & 31;
    int kb0 = (nbase < 0) ? ((-nbase) >> 5) : 0;
    int kb1 = KBLK;
    { const int mx = (NPTS - nbase) >> 5; if (mx < kb1) kb1 = mx; }
    const int kabs0 = nbase >> 5;

    v8f melacc[8];
#pragma unroll
    for (int i = 0; i < 8; ++i) melacc[i] = vzero8();

    h16* myspec = specb + wave * 512;

    for (int fc = wave; fc < FCHUNKS; fc += NWAVES) {
        const int ft0 = fc * 2, ft1 = fc * 2 + 1;
        v8f re0 = vzero8(), re1 = vzero8(), im0 = vzero8(), im1 = vzero8();
        const h16* c0 = cosB + (size_t)ft0 * 512 + lane * 16;
        const h16* c1 = cosB + (size_t)ft1 * 512 + lane * 16;
        const h16* s0 = sinB + (size_t)ft0 * 512 + lane * 16;
        const h16* s1 = sinB + (size_t)ft1 * 512 + lane * 16;

        for (int kb = kb0; kb < kb1; ++kb) {
            const size_t boff = (size_t)(kabs0 + kb) * (FTILES * 512);
            const v16h a   = *(const v16h*)(Ash + kb * 512 + lane * 16);
            const v16h bc0 = *(const v16h*)(c0 + boff);
            const v16h bs0 = *(const v16h*)(s0 + boff);
            const v16h bc1 = *(const v16h*)(c1 + boff);
            const v16h bs1 = *(const v16h*)(s1 + boff);
            if (kb + 1 < kb1) {                      // stream-ahead hint
                const size_t bo2 = boff + FTILES * 512;
                __builtin_prefetch(c0 + bo2, 0, 1);
                __builtin_prefetch(s0 + bo2, 0, 1);
            }
            re0 = wmma_f16(a, bc0, re0);
            im0 = wmma_f16(a, bs0, im0);
            re1 = wmma_f16(a, bc1, re1);
            im1 = wmma_f16(a, bs1, im1);
        }

        const v8f sp0 = re0 * re0 + im0 * im0;       // power spectrum
        const v8f sp1 = re1 * re1 + im1 * im1;

        // restage spec (C layout -> canonical [frame][32 f]) in wave-private LDS
        {
            const int fcol = lane & 15;
            const int mb = (lane >= 16) ? 8 : 0;
#pragma unroll
            for (int r = 0; r < 8; ++r) {
                myspec[(mb + r) * 32 + fcol]      = (h16)sp0[r];
                myspec[(mb + r) * 32 + 16 + fcol] = (h16)sp1[r];
            }
        }
        asm volatile("s_wait_dscnt 0" ::: "memory");  // wave-internal LDS RAW

        // gather spec as A operand (16 frames x 32 freqs)
        v16h sa;
        {
            const int M = lane & 15;
            const int dlt = (lane >= 16) ? 16 : 0;
            const v8h lo = *(const v8h*)(myspec + M * 32 + dlt);
            const v8h hi = *(const v8h*)(myspec + M * 32 + 8 + dlt);
#pragma unroll
            for (int i = 0; i < 8; ++i) { sa[i] = lo[i]; sa[8 + i] = hi[i]; }
        }

        // mel contraction: mel[16 frames x 128 mels] += spec @ fb
        const h16* fbp = fbsw + (size_t)fc * 8 * 512 + lane * 16;
#pragma unroll
        for (int mt = 0; mt < 8; ++mt) {
            const v16h fbt = *(const v16h*)(fbp + mt * 512);
            melacc[mt] = wmma_f16(sa, fbt, melacc[mt]);
        }
    }

    // phase 4: deterministic cross-wave reduction into LDS mel buffer
    for (int w = 0; w < NWAVES; ++w) {
        if (wave == w) {
            const int col = lane & 15;
            const int mrow = (lane >= 16) ? 8 : 0;
#pragma unroll
            for (int mt = 0; mt < 8; ++mt)
#pragma unroll
                for (int r = 0; r < 8; ++r)
                    melb[(mrow + r) * NMELS + mt * 16 + col] += melacc[mt][r];
        }
        __syncthreads();
    }

    // store [B,1,M,T]
    for (int e = tid; e < 16 * NMELS; e += 256) {
        const int frame = e >> 7, m = e & 127;
        const int t = t0 + frame;
        if (t < TOUT)
            out[((size_t)b * NMELS + m) * TOUT + t] = melb[e];
    }
}

// ---------------------------------------------------------------------------
extern "C" void kernel_launch(void* const* d_in, const int* in_sizes, int n_in,
                              void* d_out, int out_size, void* d_ws, size_t ws_size,
                              hipStream_t stream) {
    (void)in_sizes; (void)n_in; (void)out_size; (void)ws_size;
    const float* x     = (const float*)d_in[0];
    const float* lambd = (const float*)d_in[1];
    float* out = (float*)d_out;
    char*  ws  = (char*)d_ws;

    float* xc   = (float*)(ws + XC_OFF);
    h16*   cosB = (h16*)(ws + COS_OFF);
    h16*   sinB = (h16*)(ws + SIN_OFF);
    h16*   fbsw = (h16*)(ws + FB_OFF);

    // Allow >64KB dynamic LDS (CDNA5 WGP has 320KB). Best-effort, idempotent.
    (void)hipFuncSetAttribute((const void*)melgab_main,
                              hipFuncAttributeMaxDynamicSharedMemorySize,
                              SMEM_BYTES);

    melgab_center<<<BATCH, 256, 0, stream>>>(x, xc);
    melgab_trig<<<KABS * FTILES, 256, 0, stream>>>(cosB, sinB);
    melgab_fb<<<(FB_ELE + 255) / 256, 256, 0, stream>>>(fbsw);
    melgab_main<<<BATCH * TT, 256, SMEM_BYTES, stream>>>(xc, cosB, sinB, fbsw,
                                                         lambd, out);
}
